// cls_model_17910013625104
// MI455X (gfx1250) — matrix operand order repaired and checked by reading, not compiled, not run on hardware
//
#include <hip/hip_runtime.h>
#include <hip/hip_bf16.h>

#define BATCH 32
#define NN    1024

typedef __bf16 bf16_t;
typedef __attribute__((ext_vector_type(16))) __bf16 v16bf;
typedef __attribute__((ext_vector_type(8)))  __bf16 v8bf;
typedef __attribute__((ext_vector_type(8)))  float  v8f;

__device__ __forceinline__ v8f wmma_bf16(v16bf a, v16bf b, v8f c) {
    return __builtin_amdgcn_wmma_f32_16x16x32_bf16(false, a, false, b,
                                                   (short)0, c, false, false);
}

__device__ __forceinline__ v16bf cat8(v8bf lo, v8bf hi) {
    return __builtin_shufflevector(lo, hi, 0, 1, 2, 3, 4, 5, 6, 7,
                                   8, 9, 10, 11, 12, 13, 14, 15);
}

// ---------------------------------------------------------------------------
// A fragment (16x32 bf16): lane(M = lane&15, half = lane>>4), element j ->
//   K = k0 + half*16 + j
// => one contiguous run of 16 at k0+half*16.
// ---------------------------------------------------------------------------
template <int F>
__device__ __forceinline__ v16bf load_a_f32_t(const float* __restrict__ tile,
                                              int mrow, int half, int k0) {
    v16bf a;
    if constexpr (F % 32 == 0) {
        const float* row = tile + (size_t)mrow * F + k0 + half * 16;
        float4 a0 = *(const float4*)(row + 0);
        float4 a1 = *(const float4*)(row + 4);
        float4 b0 = *(const float4*)(row + 8);
        float4 b1 = *(const float4*)(row + 12);
        a[0] = (bf16_t)a0.x;  a[1] = (bf16_t)a0.y;  a[2] = (bf16_t)a0.z;  a[3] = (bf16_t)a0.w;
        a[4] = (bf16_t)a1.x;  a[5] = (bf16_t)a1.y;  a[6] = (bf16_t)a1.z;  a[7] = (bf16_t)a1.w;
        a[8] = (bf16_t)b0.x;  a[9] = (bf16_t)b0.y;  a[10] = (bf16_t)b0.z; a[11] = (bf16_t)b0.w;
        a[12] = (bf16_t)b1.x; a[13] = (bf16_t)b1.y; a[14] = (bf16_t)b1.z; a[15] = (bf16_t)b1.w;
    } else {
        // small-F path: unconditional clamped load + value mask (no branches)
        const float* row = tile + (size_t)mrow * F;
#pragma unroll
        for (int j = 0; j < 16; ++j) {
            int k  = k0 + half * 16 + j;
            int kc = (k < F) ? k : 0;
            float v = row[kc];
            v = (k < F) ? v : 0.0f;
            a[j] = (bf16_t)v;
        }
    }
    return a;
}

// A fragment from bf16 row-major adjacency (ld = NN): two aligned 16B loads
__device__ __forceinline__ v16bf load_a_adj(const bf16_t* __restrict__ tile,
                                            int mrow, int half, int k0) {
    const bf16_t* row = tile + (size_t)mrow * NN + k0 + half * 16;
    v8bf lo = *(const v8bf*)(row);
    v8bf hi = *(const v8bf*)(row + 8);
    return cat8(lo, hi);
}

// B fragment (32x16): lane(n = lane&15, half), element j -> B[half*16 + j, n].
// Gram kernel: B[k,n] = x[col, k] -> contiguous 16-float run per lane.
template <int F>
__device__ __forceinline__ v16bf load_b_gram(const float* __restrict__ xrow,
                                             int k0, int half) {
    v16bf b;
    if constexpr (F % 32 == 0) {
        const float* p = xrow + k0 + half * 16;
        float4 c0 = *(const float4*)(p + 0);
        float4 c1 = *(const float4*)(p + 4);
        float4 c2 = *(const float4*)(p + 8);
        float4 c3 = *(const float4*)(p + 12);
        b[0] = (bf16_t)c0.x;  b[1] = (bf16_t)c0.y;  b[2] = (bf16_t)c0.z;  b[3] = (bf16_t)c0.w;
        b[4] = (bf16_t)c1.x;  b[5] = (bf16_t)c1.y;  b[6] = (bf16_t)c1.z;  b[7] = (bf16_t)c1.w;
        b[8] = (bf16_t)c2.x;  b[9] = (bf16_t)c2.y;  b[10] = (bf16_t)c2.z; b[11] = (bf16_t)c2.w;
        b[12] = (bf16_t)c3.x; b[13] = (bf16_t)c3.y; b[14] = (bf16_t)c3.z; b[15] = (bf16_t)c3.w;
    } else {
#pragma unroll
        for (int j = 0; j < 16; ++j) {
            int k  = k0 + half * 16 + j;
            int kc = (k < F) ? k : 0;
            float v = xrow[kc];
            v = (k < F) ? v : 0.0f;
            b[j] = (bf16_t)v;
        }
    }
    return b;
}

// ---------------------------------------------------------------------------
// 1) per-row squared norms (one wave per row)
// ---------------------------------------------------------------------------
__global__ void rowsq_kernel(const float* __restrict__ x, float* __restrict__ sq, int F) {
    int row  = (blockIdx.x * blockDim.x + threadIdx.x) >> 5;
    int lane = threadIdx.x & 31;
    if (row >= BATCH * NN) return;
    const float* p = x + (size_t)row * F;
    float s = 0.f;
    for (int f = lane; f < F; f += 32) { float v = p[f]; s += v * v; }
#pragma unroll
    for (int off = 16; off; off >>= 1) s += __shfl_down(s, off, 32);
    if (lane == 0) sq[row] = s;
}

// ---------------------------------------------------------------------------
// 2) adjacency:  adj[b,n,m] = exp(2*<x_n,x_m> - sq_n - sq_m)  (bf16 out)
//    wave computes 32x32 (2x2 tiles): A fragments reused across 2 B fragments
// ---------------------------------------------------------------------------
template <int F>
__global__ void gram_exp_kernel(const float* __restrict__ x, const float* __restrict__ sq,
                                bf16_t* __restrict__ adj) {
    int wave = threadIdx.x >> 5, lane = threadIdx.x & 31;
    int half = lane >> 4, nloc = lane & 15;
    int b    = blockIdx.z;
    int row0 = blockIdx.y * 32;
    int col0 = blockIdx.x * 128 + wave * 32;
    const float* xb    = x + (size_t)b * NN * F;
    const float* xrow  = xb + (size_t)row0 * F;
    const float* xcol0 = xb + (size_t)(col0 + nloc) * F;
    const float* xcol1 = xb + (size_t)(col0 + 16 + nloc) * F;
    v8f c00 = {}, c01 = {}, c10 = {}, c11 = {};
#pragma unroll 1
    for (int k0 = 0; k0 < F; k0 += 32) {
        v16bf a0 = load_a_f32_t<F>(xrow, nloc, half, k0);
        v16bf a1 = load_a_f32_t<F>(xrow + (size_t)16 * F, nloc, half, k0);
        v16bf b0 = load_b_gram<F>(xcol0, k0, half);
        v16bf b1 = load_b_gram<F>(xcol1, k0, half);
        c00 = wmma_bf16(a0, b0, c00);
        c01 = wmma_bf16(a0, b1, c01);
        c10 = wmma_bf16(a1, b0, c10);
        c11 = wmma_bf16(a1, b1, c11);
    }
    const float* sqb = sq + b * NN;
    bf16_t* out = adj + (size_t)b * NN * NN;
    float sqc0 = sqb[col0 + nloc];
    float sqc1 = sqb[col0 + 16 + nloc];
    auto emit = [&](v8f acc, int rbase, int cbase, float sqc) {
#pragma unroll
        for (int r = 0; r < 8; ++r) {
            int row = rbase + r + half * 8;
            float v = __expf(2.0f * acc[r] - sqb[row] - sqc);
            out[(size_t)row * NN + cbase + nloc] = (bf16_t)v;
        }
    };
    emit(c00, row0,      col0,      sqc0);
    emit(c01, row0,      col0 + 16, sqc1);
    emit(c10, row0 + 16, col0,      sqc0);
    emit(c11, row0 + 16, col0 + 16, sqc1);
}

// ---------------------------------------------------------------------------
// 3) degree -> dis = rsqrt(rowsum(adj))   (one wave per row)
// ---------------------------------------------------------------------------
__global__ void degree_kernel(const bf16_t* __restrict__ adj, float* __restrict__ dis) {
    int row  = (blockIdx.x * blockDim.x + threadIdx.x) >> 5;
    int lane = threadIdx.x & 31;
    if (row >= BATCH * NN) return;
    const bf16_t* p = adj + (size_t)row * NN;
    float s = 0.f;
    for (int m = lane; m < NN; m += 32) s += (float)p[m];
#pragma unroll
    for (int off = 16; off; off >>= 1) s += __shfl_down(s, off, 32);
    if (lane == 0) dis[row] = rsqrtf(s);
}

// ---------------------------------------------------------------------------
// 4) Laplacian apply: y = alpha*(xin - dis.*(A @ (dis.*xin))) + beta*xprev
//    Block tile 32n x 128f; wave = 32x32 (2x2) register block.
//    LDS tile transposed [f][m]: B-fragment reads are 2x ds_load_b128.
// ---------------------------------------------------------------------------
template <int F>
__global__ void lapply_kernel(const bf16_t* __restrict__ adj, const float* __restrict__ dis,
                              const float* __restrict__ xin, const float* __restrict__ xprev,
                              float* __restrict__ y, float alpha, float beta) {
    __shared__ bf16_t xt[128][40];  // [f][m], 80B pitch (16B-aligned chunks)
    int tid = threadIdx.x, wave = tid >> 5, lane = tid & 31;
    int half = lane >> 4, nloc = lane & 15;
    int b  = blockIdx.z;
    int n0 = blockIdx.y * 32;
    int f0 = blockIdx.x * 128;
    const bf16_t* adjb = adj + (size_t)b * NN * NN + (size_t)n0 * NN;
    const float*  disb = dis + b * NN;
    const float*  xinb = xin + (size_t)b * NN * F;

    int sf  = f0 + tid;          // staging: thread owns one f row of the tile
    int sfc = (sf < F) ? sf : 0;
    bool sfv = (sf < F);
    int fw = wave * 32;          // wave's 32-col window within the 128-f tile

    v8f c00 = {}, c01 = {}, c10 = {}, c11 = {};
#pragma unroll 1
    for (int m0 = 0; m0 < NN; m0 += 32) {
        __syncthreads();
#pragma unroll
        for (int g = 0; g < 4; ++g) {
            v8bf v;
#pragma unroll
            for (int mi = 0; mi < 8; ++mi) {
                int m = m0 + g * 8 + mi;
                float t = disb[m] * xinb[(size_t)m * F + sfc];
                v[mi] = (bf16_t)(sfv ? t : 0.0f);
            }
            *(v8bf*)&xt[tid][g * 8] = v;
        }
        __syncthreads();
        v16bf a0 = load_a_adj(adjb, nloc,      half, m0);
        v16bf a1 = load_a_adj(adjb, nloc + 16, half, m0);
        int cc0 = fw + nloc, cc1 = fw + 16 + nloc;
        v16bf b0 = cat8(*(const v8bf*)&xt[cc0][half * 16],
                        *(const v8bf*)&xt[cc0][half * 16 + 8]);
        v16bf b1 = cat8(*(const v8bf*)&xt[cc1][half * 16],
                        *(const v8bf*)&xt[cc1][half * 16 + 8]);
        c00 = wmma_bf16(a0, b0, c00);
        c01 = wmma_bf16(a0, b1, c01);
        c10 = wmma_bf16(a1, b0, c10);
        c11 = wmma_bf16(a1, b1, c11);
    }
    float* yb = y + (size_t)b * NN * F;
    const float* xpb = xprev ? (xprev + (size_t)b * NN * F) : nullptr;
    auto emit = [&](v8f acc, int rb, int cb) {
        int fg = f0 + cb + nloc;
        if (fg < F) {
#pragma unroll
            for (int r = 0; r < 8; ++r) {
                int n = n0 + rb + r + half * 8;
                float lx  = xinb[(size_t)n * F + fg] - disb[n] * acc[r];
                float o   = alpha * lx + (xpb ? beta * xpb[(size_t)n * F + fg] : 0.f);
                yb[(size_t)n * F + fg] = o;
            }
        }
    };
    emit(c00, 0,  fw);      emit(c01, 0,  fw + 16);
    emit(c10, 16, fw);      emit(c11, 16, fw + 16);
}

// ---------------------------------------------------------------------------
// 5) Chebyshev weight GEMM + bias + ReLU over rows r in [0, B*N)
//    Block tile 32r x 128g; wave = 32x32 (2x2) register block.
// ---------------------------------------------------------------------------
template <int F, int G>
__global__ void chebmm_kernel(const float* __restrict__ x0, const float* __restrict__ x1,
                              const float* __restrict__ x2, const float* __restrict__ W,
                              const float* __restrict__ bias, float* __restrict__ out) {
    __shared__ bf16_t wt[128][40];  // [g][f] transposed
    int tid = threadIdx.x, wave = tid >> 5, lane = tid & 31;
    int half = lane >> 4, nloc = lane & 15;
    int r0 = blockIdx.y * 32;
    int g0 = blockIdx.x * 128;

    int sg = g0 + tid;            // staging: thread owns one g row (always < G)
    int gw = wave * 32;

    constexpr int FCH = (F + 31) / 32;  // K chunks per k-order
    v8f c00 = {}, c01 = {}, c10 = {}, c11 = {};
#pragma unroll 1
    for (int kc = 0; kc < 3 * FCH; ++kc) {
        int k  = kc / FCH;
        int f0 = (kc % FCH) * 32;
        const float* xs = (k == 0) ? x0 : (k == 1) ? x1 : x2;
        const float* Wk = W + (size_t)k * F * G;
        __syncthreads();
#pragma unroll
        for (int g = 0; g < 4; ++g) {
            v8bf v;
#pragma unroll
            for (int fi = 0; fi < 8; ++fi) {
                int f  = f0 + g * 8 + fi;
                int fc = (f < F) ? f : 0;
                float t = Wk[(size_t)fc * G + sg];
                v[fi] = (bf16_t)((f < F) ? t : 0.0f);
            }
            *(v8bf*)&wt[tid][g * 8] = v;
        }
        __syncthreads();
        v16bf a0 = load_a_f32_t<F>(xs + (size_t)r0 * F,        nloc, half, f0);
        v16bf a1 = load_a_f32_t<F>(xs + (size_t)(r0 + 16) * F, nloc, half, f0);
        int cc0 = gw + nloc, cc1 = gw + 16 + nloc;
        v16bf b0 = cat8(*(const v8bf*)&wt[cc0][half * 16],
                        *(const v8bf*)&wt[cc0][half * 16 + 8]);
        v16bf b1 = cat8(*(const v8bf*)&wt[cc1][half * 16],
                        *(const v8bf*)&wt[cc1][half * 16 + 8]);
        c00 = wmma_bf16(a0, b0, c00);
        c01 = wmma_bf16(a0, b1, c01);
        c10 = wmma_bf16(a1, b0, c10);
        c11 = wmma_bf16(a1, b1, c11);
    }
    auto emit = [&](v8f acc, int rb, int cb) {
        int g = g0 + cb + nloc;
        float bv = bias[g];
#pragma unroll
        for (int r = 0; r < 8; ++r) {
            int row = r0 + rb + r + half * 8;
            out[(size_t)row * G + g] = fmaxf(acc[r] + bv, 0.f);
        }
    };
    emit(c00, 0,  gw);      emit(c01, 0,  gw + 16);
    emit(c10, 16, gw);      emit(c11, 16, gw + 16);
}

// ---------------------------------------------------------------------------
// 6) global max pool over nodes
// ---------------------------------------------------------------------------
__global__ void maxpool_kernel(const float* __restrict__ x, float* __restrict__ y) {
    int b = blockIdx.y;
    int g = blockIdx.x * blockDim.x + threadIdx.x;
    const float* p = x + (size_t)b * NN * 1024 + g;
    float m = -3.4e38f;
    for (int n = 0; n < NN; ++n) m = fmaxf(m, p[(size_t)n * 1024]);
    y[b * 1024 + g] = m;
}

// ---------------------------------------------------------------------------
// 7) small FC layers (32 rows -- trivial)
// ---------------------------------------------------------------------------
__global__ void fc_kernel(const float* __restrict__ x, const float* __restrict__ w,
                          const float* __restrict__ bias, float* __restrict__ y,
                          int Kdim, int Cdim, int do_relu) {
    int idx = blockIdx.x * blockDim.x + threadIdx.x;
    if (idx >= 32 * Cdim) return;
    int r = idx / Cdim, c = idx % Cdim;
    const float* xr = x + (size_t)r * Kdim;
    float s = bias[c];
    for (int k = 0; k < Kdim; ++k) s += xr[k] * w[(size_t)k * Cdim + c];
    y[idx] = do_relu ? fmaxf(s, 0.f) : s;
}

// ---------------------------------------------------------------------------
template <int F, int G>
static void run_layer(const float* xin, const float* W, const float* bias, float* out,
                      bf16_t* adj, float* dis, float* sq, float* xb1, float* xb2,
                      hipStream_t stream) {
    dim3 blk128(128);
    const int waveBlocks = (BATCH * NN) / 8;
    rowsq_kernel<<<waveBlocks, 256, 0, stream>>>(xin, sq, F);
    gram_exp_kernel<F><<<dim3(NN / 128, NN / 32, BATCH), blk128, 0, stream>>>(xin, sq, adj);
    degree_kernel<<<waveBlocks, 256, 0, stream>>>(adj, dis);
    const int fx = (F + 127) / 128;
    lapply_kernel<F><<<dim3(fx, NN / 32, BATCH), blk128, 0, stream>>>(
        adj, dis, xin, nullptr, xb1, 1.0f, 0.0f);     // x1 = L x0
    lapply_kernel<F><<<dim3(fx, NN / 32, BATCH), blk128, 0, stream>>>(
        adj, dis, xb1, xin, xb2, 2.0f, -1.0f);        // x2 = 2 L x1 - x0
    chebmm_kernel<F, G><<<dim3(G / 128, (BATCH * NN) / 32), blk128, 0, stream>>>(
        xin, xb1, xb2, W, bias, out);
}

extern "C" void kernel_launch(void* const* d_in, const int* in_sizes, int n_in,
                              void* d_out, int out_size, void* d_ws, size_t ws_size,
                              hipStream_t stream) {
    const float* x    = (const float*)d_in[0];
    const float* W1   = (const float*)d_in[1];
    const float* b1   = (const float*)d_in[2];
    const float* W2   = (const float*)d_in[3];
    const float* b2   = (const float*)d_in[4];
    const float* W3   = (const float*)d_in[5];
    const float* b3   = (const float*)d_in[6];
    const float* fc1w = (const float*)d_in[7];
    const float* fc1b = (const float*)d_in[8];
    const float* fc2w = (const float*)d_in[9];
    const float* fc2b = (const float*)d_in[10];
    const float* fc3w = (const float*)d_in[11];
    const float* fc3b = (const float*)d_in[12];

    char* wp = (char*)d_ws;
    auto take = [&](size_t bytes) -> void* {
        void* p = (void*)wp;
        wp += (bytes + 255) & ~(size_t)255;
        return p;
    };
    bf16_t* adj = (bf16_t*)take((size_t)BATCH * NN * NN * sizeof(bf16_t));  // 64 MB
    float* dis  = (float*)take((size_t)BATCH * NN * sizeof(float));
    float* sq   = (float*)take((size_t)BATCH * NN * sizeof(float));
    float* xb1  = (float*)take((size_t)BATCH * NN * 512 * sizeof(float));
    float* xb2  = (float*)take((size_t)BATCH * NN * 512 * sizeof(float));
    float* out1 = (float*)take((size_t)BATCH * NN * 128 * sizeof(float));
    float* out2 = (float*)take((size_t)BATCH * NN * 512 * sizeof(float));
    float* out3 = (float*)take((size_t)BATCH * NN * 1024 * sizeof(float));
    float* pool = (float*)take((size_t)BATCH * 1024 * sizeof(float));
    float* fco1 = (float*)take((size_t)BATCH * 512 * sizeof(float));
    float* fco2 = (float*)take((size_t)BATCH * 128 * sizeof(float));
    (void)in_sizes; (void)n_in; (void)out_size; (void)ws_size;

    run_layer<6, 128>(x, W1, b1, out1, adj, dis, sq, xb1, xb2, stream);
    run_layer<128, 512>(out1, W2, b2, out2, adj, dis, sq, xb1, xb2, stream);
    run_layer<512, 1024>(out2, W3, b3, out3, adj, dis, sq, xb1, xb2, stream);

    maxpool_kernel<<<dim3(1024 / 256, BATCH), 256, 0, stream>>>(out3, pool);
    fc_kernel<<<(32 * 512 + 255) / 256, 256, 0, stream>>>(pool, fc1w, fc1b, fco1, 1024, 512, 1);
    fc_kernel<<<(32 * 128 + 255) / 256, 256, 0, stream>>>(fco1, fc2w, fc2b, fco2, 512, 128, 1);
    fc_kernel<<<(32 * 40 + 255) / 256, 256, 0, stream>>>(fco2, fc3w, fc3b, (float*)d_out, 128, 40, 0);
}
